// WindowCrossAttention_54589034332169
// MI455X (gfx1250) — matrix operand order repaired and checked by reading, not compiled, not run on hardware
//
#include <hip/hip_runtime.h>
#include <hip/hip_bf16.h>

typedef __attribute__((ext_vector_type(16))) _Float16 v16h;
typedef __attribute__((ext_vector_type(8)))  _Float16 v8h;
typedef __attribute__((ext_vector_type(4)))  _Float16 v4h;
typedef __attribute__((ext_vector_type(8)))  float    v8f;

#define NHEAD   6
#define HDIM    32
#define CDIM    192
#define NTOK    64
#define QKSCALE 0.17677669529663687f   // 32^-0.5
#define NTHREADS 192                   // 6 waves (wave32), one wave per head

// d_ws layout (in _Float16 elements)
#define WS_QW     0                        // 192*192        f16 weights
#define WS_KVW    36864                    // 384*192
#define WS_PW     110592                   // 192*192
#define WS_BIAS   147456                   // 6*64*16*4      swizzled [h][row][lane][n]
#define WS_MASK   172032                   // 1024*64*16*4   swizzled [w][row][lane][n]
#define WS_TOTAL  4366336                  // halves (~8.7 MB)

// ---------------------------------------------------------------------------
// Fragment loaders for v_wmma_f32_16x16x32_f16 (wave32).
// A (16x32, f16, row-major source M x K, ld in halves):
//   lane l: row = r0 + (l&15); lanes 0-15 hold K=[k0..k0+15],
//   lanes 16-31 hold K=[k0+16..k0+31]  (matches load_b_frag)
__device__ __forceinline__ v16h load_a_frag(const _Float16* src, int ld,
                                            int r0, int k0, int lane) {
  const int hl = lane >> 4;
  const int r  = r0 + (lane & 15);
  const v8h* p0 = (const v8h*)(src + r * ld + k0 + hl * 16);
  const v8h* p1 = (const v8h*)(src + r * ld + k0 + hl * 16 + 8);
  v8h x = *p0, y = *p1;
  v16h out;
#pragma unroll
  for (int i = 0; i < 8; ++i) { out[i] = x[i]; out[i + 8] = y[i]; }
  return out;
}

// B (32x16, f16) loaded from B^T stored row-major N x K (ld in halves):
//   lane l: col n = n0 + (l&15); lanes 0-15 hold K=[k0..k0+15],
//   lanes 16-31 hold K=[k0+16..k0+31], 2 halves per VGPR ascending.
__device__ __forceinline__ v16h load_b_frag(const _Float16* srcT, int ld,
                                            int n0, int k0, int lane) {
  const int n  = n0 + (lane & 15);
  const int kl = lane >> 4;
  return *(const v16h*)(srcT + n * ld + k0 + kl * 16);
}

__device__ __forceinline__ v8f wmma_f16(v16h a, v16h b, v8f c) {
  return __builtin_amdgcn_wmma_f32_16x16x32_f16(false, a, false, b,
                                                (short)0, c, false, false);
}

// C/D layout: VGPR i, lanes 0-15 -> (M = r0+i,   N = n0+lane),
//             lanes 16-31        -> (M = r0+i+8, N = n0+lane-16)
__device__ __forceinline__ void store_c_f16(v8f c, float bias, float scale,
                                            _Float16* dst, int ld,
                                            int r0, int c0, int lane) {
  const int hl = lane >> 4;
  const int cc = c0 + (lane & 15);
#pragma unroll
  for (int i = 0; i < 8; ++i)
    dst[(r0 + i + 8 * hl) * ld + cc] = (_Float16)((c[i] + bias) * scale);
}

__device__ __forceinline__ void store_c_f16_T(v8f c, float bias, _Float16* dst,
                                              int ldT, int r0, int c0, int lane) {
  const int hl = lane >> 4;
  const int cc = c0 + (lane & 15);
#pragma unroll
  for (int i = 0; i < 8; ++i)
    dst[cc * ldT + (r0 + i + 8 * hl)] = (_Float16)(c[i] + bias);
}

// ---------------------------------------------------------------------------
// Prep 1: f32 weights -> f16; build swizzled rel-pos bias [h][row][lane][n].
__global__ void wca_prep_kernel(const float* __restrict__ qw,
                                const float* __restrict__ kvw,
                                const float* __restrict__ pw,
                                const float* __restrict__ table,
                                _Float16* __restrict__ ws) {
  int idx = blockIdx.x * 256 + threadIdx.x;
  if (idx < 36864) {
    ws[WS_QW + idx] = (_Float16)qw[idx];
  } else if (idx < 110592) {
    ws[idx] = (_Float16)kvw[idx - 36864];
  } else if (idx < 147456) {
    ws[idx] = (_Float16)pw[idx - 110592];
  } else if (idx < WS_MASK) {
    int t = idx - WS_BIAS;
    int n2 = t & 3, lc = (t >> 2) & 15, row = (t >> 6) & 63, h = t >> 12;
    int col = n2 * 16 + lc;
    int nh = row >> 3, nw = row & 7, mh = col >> 3, mw = col & 7;
    int rel = (nh - mh + 7) * 15 + (nw - mw + 7);
    ws[idx] = (_Float16)table[rel * NHEAD + h];
  }
}

// Prep 2: swizzle window mask -> f16 [w][row][lane][n] (L2-resident, 8 MB).
__global__ void wca_mask_kernel(const float* __restrict__ mask,
                                _Float16* __restrict__ ws) {
  int o = blockIdx.x * 256 + threadIdx.x;            // 1024*64*16*4
  int n2 = o & 3, lc = (o >> 2) & 15, row = (o >> 6) & 63, w = o >> 12;
  ws[WS_MASK + o] = (_Float16)mask[(size_t)w * 4096 + row * 64 + n2 * 16 + lc];
}

// ---------------------------------------------------------------------------
// Main: one block per window, 6 waves; wave w = head w in attention phase.
__global__ __launch_bounds__(NTHREADS)
void wca_main_kernel(const float* __restrict__ x1, const float* __restrict__ x2,
                     const float* __restrict__ qb, const float* __restrict__ kvb,
                     const float* __restrict__ pb,
                     const _Float16* __restrict__ ws,
                     float* __restrict__ out) {
  // LDS: 73728 halves = 144 KB -> 2 blocks per WGP
  __shared__ __align__(32) _Float16 smem[73728];
  _Float16* sX1   = smem;             // 64x192 (phase 1-2)
  _Float16* sX2   = smem + 12288;     // 64x192 (phase 1-2)
  _Float16* sAttn = smem;             // 6 x 64x64 (phase 3, aliases sX1/sX2)
  float*    sOut  = (float*)smem;     // 64x192 f32 (phase 4, aliases attn)
  _Float16* sQ    = smem + 24576;     // 64x192  (pre-scaled by QKSCALE)
  _Float16* sK    = smem + 36864;     // 64x192
  _Float16* sVt   = smem + 49152;     // 192x64  (v transposed: dim-major)
  _Float16* sO    = smem + 61440;     // 64x192

  const _Float16* qwh    = ws + WS_QW;
  const _Float16* kvwh   = ws + WS_KVW;
  const _Float16* pwh    = ws + WS_PW;
  const _Float16* biasSw = ws + WS_BIAS;
  const _Float16* maskSw = ws + WS_MASK;

  const int tid  = threadIdx.x;
  const int lane = tid & 31;
  const int wave = tid >> 5;                 // 0..5
  const int b    = blockIdx.x;
  const float* xg1 = x1 + (size_t)b * 12288;
  const float* xg2 = x2 + (size_t)b * 12288;

  // ---- Phase 1: stage x1,x2 into LDS as f16 ------------------------------
  for (int i = tid * 4; i < 12288; i += NTHREADS * 4) {
    float4 a = *(const float4*)(xg1 + i);
    sX1[i + 0] = (_Float16)a.x; sX1[i + 1] = (_Float16)a.y;
    sX1[i + 2] = (_Float16)a.z; sX1[i + 3] = (_Float16)a.w;
    float4 c = *(const float4*)(xg2 + i);
    sX2[i + 0] = (_Float16)c.x; sX2[i + 1] = (_Float16)c.y;
    sX2[i + 2] = (_Float16)c.z; sX2[i + 3] = (_Float16)c.w;
  }
  __syncthreads();

  // ---- Phase 2: q/k/v projections, 3 uniform loops of 8 tiles per wave ---
#pragma unroll 1
  for (int j = 0; j < 8; ++j) {            // Q: 48 tiles
    const int t = wave * 8 + j;
    const int mi = t / 12, ni = t % 12;
    v8f acc = {0.f, 0.f, 0.f, 0.f, 0.f, 0.f, 0.f, 0.f};
#pragma unroll
    for (int k0 = 0; k0 < CDIM; k0 += 32)
      acc = wmma_f16(load_a_frag(sX1, CDIM, mi * 16, k0, lane),
                     load_b_frag(qwh, CDIM, ni * 16, k0, lane), acc);
    store_c_f16(acc, qb[ni * 16 + (lane & 15)], QKSCALE,
                sQ, CDIM, mi * 16, ni * 16, lane);
  }
#pragma unroll 1
  for (int j = 0; j < 8; ++j) {            // K: 48 tiles
    const int t = wave * 8 + j;
    const int mi = t / 12, ni = t % 12;
    v8f acc = {0.f, 0.f, 0.f, 0.f, 0.f, 0.f, 0.f, 0.f};
#pragma unroll
    for (int k0 = 0; k0 < CDIM; k0 += 32)
      acc = wmma_f16(load_a_frag(sX2, CDIM, mi * 16, k0, lane),
                     load_b_frag(kvwh, CDIM, ni * 16, k0, lane), acc);
    store_c_f16(acc, kvb[ni * 16 + (lane & 15)], 1.0f,
                sK, CDIM, mi * 16, ni * 16, lane);
  }
#pragma unroll 1
  for (int j = 0; j < 8; ++j) {            // V: 48 tiles, transposed store
    const int t = wave * 8 + j;
    const int mi = t / 12, ni = t % 12;
    v8f acc = {0.f, 0.f, 0.f, 0.f, 0.f, 0.f, 0.f, 0.f};
#pragma unroll
    for (int k0 = 0; k0 < CDIM; k0 += 32)
      acc = wmma_f16(load_a_frag(sX2, CDIM, mi * 16, k0, lane),
                     load_b_frag(kvwh + 192 * 192, CDIM, ni * 16, k0, lane), acc);
    store_c_f16_T(acc, kvb[192 + ni * 16 + (lane & 15)],
                  sVt, NTOK, mi * 16, ni * 16, lane);
  }
  __syncthreads();

  // ---- Phase 3: per-head attention (head = wave) -------------------------
  {
    const int h = wave;
    const int lc = lane & 15, hl = lane >> 4;
    const _Float16* biasH = biasSw + ((h * 64) * 16 + lc) * 4;
    const _Float16* maskW = maskSw + (((size_t)(b & 1023) * 64) * 16 + lc) * 4;
    _Float16*       attnH = sAttn + h * 4096;

    // QK^T (+bias+mask; q already scaled), softmax, store f16 attn to LDS
    for (int m = 0; m < 4; ++m) {
      v16h aq = load_a_frag(sQ, CDIM, m * 16, h * HDIM, lane);
      v8f c[4];
#pragma unroll
      for (int n = 0; n < 4; ++n) {
        v16h bk = load_b_frag(sK, CDIM, n * 16, h * HDIM, lane);
        v8f z = {0.f, 0.f, 0.f, 0.f, 0.f, 0.f, 0.f, 0.f};
        c[n] = wmma_f16(aq, bk, z);
      }
#pragma unroll
      for (int i = 0; i < 8; ++i) {
        const int row = m * 16 + i + 8 * hl;
        v4h bb = *(const v4h*)(biasH + row * 64);   // [row][lane][0..3]
        v4h mm = *(const v4h*)(maskW + row * 64);
        float mx = -3.0e30f;
#pragma unroll
        for (int n = 0; n < 4; ++n) {
          float v = c[n][i] + (float)bb[n] + (float)mm[n];
          c[n][i] = v;
          mx = fmaxf(mx, v);
        }
#pragma unroll
        for (int s = 1; s < 16; s <<= 1) mx = fmaxf(mx, __shfl_xor(mx, s));
        float sum = 0.f;
#pragma unroll
        for (int n = 0; n < 4; ++n) {
          float e = __expf(c[n][i] - mx);
          c[n][i] = e;
          sum += e;
        }
#pragma unroll
        for (int s = 1; s < 16; s <<= 1) sum += __shfl_xor(sum, s);
        const float inv = 1.0f / sum;
#pragma unroll
        for (int n = 0; n < 4; ++n)
          attnH[row * 64 + n * 16 + lc] = (_Float16)(c[n][i] * inv);
      }
    }

    // AV: (64x64 attn) x (64x32 v) via v^T, two K-steps per tile
    for (int m = 0; m < 4; ++m) {
      v16h a0 = load_a_frag(attnH, NTOK, m * 16, 0, lane);
      v16h a1 = load_a_frag(attnH, NTOK, m * 16, 32, lane);
#pragma unroll
      for (int n = 0; n < 2; ++n) {
        v8f acc = {0.f, 0.f, 0.f, 0.f, 0.f, 0.f, 0.f, 0.f};
        acc = wmma_f16(a0, load_b_frag(sVt, NTOK, h * HDIM + n * 16, 0, lane), acc);
        acc = wmma_f16(a1, load_b_frag(sVt, NTOK, h * HDIM + n * 16, 32, lane), acc);
        store_c_f16(acc, 0.f, 1.0f, sO, CDIM, m * 16, h * HDIM + n * 16, lane);
      }
    }
  }
  __syncthreads();

  // ---- Phase 4: output projection (48 tiles, 8 per wave) -> LDS f32 ------
#pragma unroll 1
  for (int j = 0; j < 8; ++j) {
    const int t = wave * 8 + j;
    const int mi = t / 12, ni = t % 12;
    v8f acc = {0.f, 0.f, 0.f, 0.f, 0.f, 0.f, 0.f, 0.f};
#pragma unroll
    for (int k0 = 0; k0 < CDIM; k0 += 32)
      acc = wmma_f16(load_a_frag(sO, CDIM, mi * 16, k0, lane),
                     load_b_frag(pwh, CDIM, ni * 16, k0, lane), acc);
    const int cc = ni * 16 + (lane & 15);
    const int hl = lane >> 4;
    const float bias = pb[cc];
#pragma unroll
    for (int i = 0; i < 8; ++i)
      sOut[(mi * 16 + i + 8 * hl) * CDIM + cc] = acc[i] + bias;
  }
  __syncthreads();

  // ---- Phase 5: coalesced 128-bit store-out ------------------------------
  float* og = out + (size_t)b * 12288;
  for (int i = tid * 4; i < 12288; i += NTHREADS * 4) {
    float4 r = *(const float4*)(sOut + i);
    *(float4*)(og + i) = r;
  }
}

// ---------------------------------------------------------------------------
extern "C" void kernel_launch(void* const* d_in, const int* in_sizes, int n_in,
                              void* d_out, int out_size, void* d_ws, size_t ws_size,
                              hipStream_t stream) {
  const float* x1    = (const float*)d_in[0];
  const float* x2    = (const float*)d_in[1];
  const float* mask  = (const float*)d_in[2];
  const float* q_w   = (const float*)d_in[3];
  const float* q_b   = (const float*)d_in[4];
  const float* kv_w  = (const float*)d_in[5];
  const float* kv_b  = (const float*)d_in[6];
  const float* pr_w  = (const float*)d_in[7];
  const float* pr_b  = (const float*)d_in[8];
  const float* table = (const float*)d_in[9];
  _Float16* ws = (_Float16*)d_ws;
  float* out = (float*)d_out;

  const int B = in_sizes[0] / (NTOK * CDIM);   // 8192

  wca_prep_kernel<<<(WS_MASK + 255) / 256, 256, 0, stream>>>(q_w, kv_w, pr_w,
                                                             table, ws);
  wca_mask_kernel<<<(1024 * 4096) / 256, 256, 0, stream>>>(mask, ws);
  wca_main_kernel<<<B, NTHREADS, 0, stream>>>(x1, x2, q_b, kv_b, pr_b, ws, out);
}
